// NCE_model_36928128811088
// MI455X (gfx1250) — matrix operand order repaired and checked by reading, not compiled, not run on hardware
//
#include <hip/hip_runtime.h>
#include <hip/hip_bf16.h>
#include <math.h>

// ---------------------------------------------------------------------------
// NCE loss, fused flash-style on gfx1250:
//   M = src @ tgt^T (8192x8192x64) via v_wmma_f32_16x16x32_f16 with an
//   online (chunk-max) logsumexp over targets of (M + logw). M is never
//   materialized; all operands (~4.3MB) live in L2.
//   Masked row-sum / pair-count factor out of the stream:
//     cnt_j  = ccount[label_j]
//     msum_j = src_feas[j] . tsum[label_j]   (exact f32, tiny precompute)
//   Hot loop: immediate-offset b128 loads + 2 WMMA + 4 pk_fma + max tree +
//   9 raw v_exp per 16-target chunk, unrolled x4 so the scheduler can fill
//   the WMMA->VALU hazard slots with independent chunks.
//   All reductions are fixed-order (no float atomics) -> deterministic.
// ---------------------------------------------------------------------------

typedef __attribute__((ext_vector_type(16))) _Float16 v16h;
typedef __attribute__((ext_vector_type(8)))  _Float16 v8h;
typedef __attribute__((ext_vector_type(8)))  float    v8f;

#define LOG2E   1.4426950408889634f
#define INV_L2E 0.6931471805599453f
#define NEG_INF (-1e30f)
#define DELTA   0.1f
#define DFEA    64   // feature dim (reference: D = 64)

static __device__ __forceinline__ v16h cat8(v8h lo, v8h hi) {
    return __builtin_shufflevector(lo, hi, 0, 1, 2, 3, 4, 5, 6, 7,
                                           8, 9, 10, 11, 12, 13, 14, 15);
}

// raw v_exp_f32 (base-2); args <= 0, underflow-to-zero wanted
static __device__ __forceinline__ float fast_exp2(float x) {
    return __builtin_amdgcn_exp2f(x);
}

// ---- prep kernels ---------------------------------------------------------

__global__ void k_zero(int* __restrict__ counts, int* __restrict__ ccount) {
    int t = threadIdx.x;
    if (t < 16) { counts[t] = 0; ccount[t] = 0; }
}

__global__ void k_counts(const int* __restrict__ labels, int* __restrict__ counts, int n) {
    int i = blockIdx.x * blockDim.x + threadIdx.x;
    if (i < n) atomicAdd(&counts[labels[i]], 1);
}

__global__ void k_cvt(const float* __restrict__ a, _Float16* __restrict__ o, int n) {
    int i = blockIdx.x * blockDim.x + threadIdx.x;
    if (i < n) o[i] = (_Float16)a[i];
}

// per-target: argmax + top-2 gap confidence -> cls (or -1); ccount[cls]++
__global__ void k_cls(const float* __restrict__ logits, int* __restrict__ cls,
                      int* __restrict__ ccount, int Bt, int C) {
    int i = blockIdx.x * blockDim.x + threadIdx.x;
    if (i >= Bt) return;
    const float* p = logits + (size_t)i * C;
    float m1 = -INFINITY, m2 = -INFINITY;
    int arg = 0;
    for (int c = 0; c < C; ++c) {
        float v = p[c];
        if (v > m1) { m2 = m1; m1 = v; arg = c; }
        else if (v > m2) { m2 = v; }
    }
    bool conf = (m1 - m2) >= DELTA;
    cls[i] = conf ? arg : -1;
    if (conf) atomicAdd(&ccount[arg], 1);
}

// per-target log2(w) (base-2 so the hot loop needs one FMA per score)
__global__ void k_metaLw(const int* __restrict__ cls, const int* __restrict__ counts,
                         float* __restrict__ metaLw, int Bt) {
    int i = blockIdx.x * blockDim.x + threadIdx.x;
    if (i >= Bt) return;
    int c = cls[i];
    float w = (c >= 0) ? (float)counts[c] : 0.0f;
    metaLw[i] = (w > 0.0f) ? log2f(w) : NEG_INF;
}

// per (class, dim): tsum[c][d] = sum over confident targets of class c
// (fixed ascending-i order -> deterministic)
__global__ void k_tsum(const float* __restrict__ tgt, const int* __restrict__ cls,
                       float* __restrict__ tsum, int Bt, int C) {
    int t = blockIdx.x * blockDim.x + threadIdx.x;
    if (t >= C * DFEA) return;
    int c = t / DFEA, d = t % DFEA;
    float s = 0.0f;
    for (int i = 0; i < Bt; ++i)
        if (cls[i] == c) s += tgt[(size_t)i * DFEA + d];
    tsum[t] = s;
}

// per src row: msum[j] = src_feas[j] . tsum[label_j]  (exact f32)
__global__ void k_msum(const float* __restrict__ src, const int* __restrict__ labels,
                       const float* __restrict__ tsum, float* __restrict__ msum, int Bs) {
    int j = blockIdx.x * blockDim.x + threadIdx.x;
    if (j >= Bs) return;
    const float4* a = (const float4*)(src + (size_t)j * DFEA);
    const float4* b = (const float4*)(tsum + (size_t)labels[j] * DFEA);
    float s = 0.0f;
#pragma unroll
    for (int q = 0; q < DFEA / 4; ++q) {
        float4 x = a[q], y = b[q];
        s += x.x * y.x + x.y * y.y + x.z * y.z + x.w * y.w;
    }
    msum[j] = s;
}

// ---- main fused kernel ----------------------------------------------------
// One wave owns 16 src rows (WMMA N) and streams all Bt targets (WMMA M) in
// 16-row chunks: 2x v_wmma_f32_16x16x32_f16 + chunk-max online logsumexp
// (9 raw v_exp per chunk). Each lane owns one src column across all 8
// accumulator VGPRs; lane pair (l, l^16) merges at the very end.
__global__ void k_main(const _Float16* __restrict__ src16,
                       const _Float16* __restrict__ tgt16,
                       const float*    __restrict__ metaLw,
                       const int*      __restrict__ src_labels,
                       const float*    __restrict__ msumArr,
                       const int*      __restrict__ ccount,
                       float*          __restrict__ contribArr,
                       int Bt) {
    const int lane = threadIdx.x & 31;
    const int wave = threadIdx.x >> 5;
    const int gw   = blockIdx.x * (blockDim.x >> 5) + wave;
    const int col  = gw * 16 + (lane & 15);      // src row handled by this lane
    const int half = lane >> 4;                   // 0: lanes 0-15, 1: lanes 16-31

    // B operand (32x16 f16): lane holds src column col; K halves {8h..8h+7, 8h+16..8h+23}.
    const v8h* ps = (const v8h*)(src16 + (size_t)col * DFEA);
    const v16h b0 = cat8(ps[half + 0], ps[half + 2]);  // K  0..31
    const v16h b1 = cat8(ps[half + 4], ps[half + 6]);  // K 32..63

    float m2e = NEG_INF;   // running max of (M + logw) in base-2 units
    float s   = 0.0f;      // sum of 2^(y - m2e)
    const int rowA = lane & 15;
    const int tOff = half * 8;   // accumulator VGPR v holds target tBase+tOff+v

    // incrementing pointers -> immediate-offset loads inside the loop
    const v8h*    pa = (const v8h*)(tgt16 + (size_t)rowA * DFEA) + half;
    const float4* pm = (const float4*)(metaLw + tOff);

#pragma unroll 4
    for (int tBase = 0; tBase < Bt; tBase += 16) {
        // A operand (16x32 f16): lane holds target row tBase+rowA,
        // K halves {8h..8h+7, 8h+16..8h+23}.
        v16h a0 = cat8(pa[0], pa[2]);
        v16h a1 = cat8(pa[4], pa[6]);

        v8f c = {};
        c = __builtin_amdgcn_wmma_f32_16x16x32_f16(false, a0, false, b0,
                                                   (short)0, c, false, false);
        c = __builtin_amdgcn_wmma_f32_16x16x32_f16(false, a1, false, b1,
                                                   (short)0, c, false, false);

        // y[v] = M*log2(e) + log2(w)
        float4 w0 = pm[0], w1 = pm[1];
        float y[8];
        y[0] = fmaf(c[0], LOG2E, w0.x);
        y[1] = fmaf(c[1], LOG2E, w0.y);
        y[2] = fmaf(c[2], LOG2E, w0.z);
        y[3] = fmaf(c[3], LOG2E, w0.w);
        y[4] = fmaf(c[4], LOG2E, w1.x);
        y[5] = fmaf(c[5], LOG2E, w1.y);
        y[6] = fmaf(c[6], LOG2E, w1.z);
        y[7] = fmaf(c[7], LOG2E, w1.w);

        // chunk max, single online rescale, 8 value exps
        float cm = fmaxf(fmaxf(fmaxf(y[0], y[1]), fmaxf(y[2], y[3])),
                         fmaxf(fmaxf(y[4], y[5]), fmaxf(y[6], y[7])));
        float nm = fmaxf(m2e, cm);
        float e0 = fast_exp2(y[0] - nm), e1 = fast_exp2(y[1] - nm);
        float e2 = fast_exp2(y[2] - nm), e3 = fast_exp2(y[3] - nm);
        float e4 = fast_exp2(y[4] - nm), e5 = fast_exp2(y[5] - nm);
        float e6 = fast_exp2(y[6] - nm), e7 = fast_exp2(y[7] - nm);
        float chunk = ((e0 + e1) + (e2 + e3)) + ((e4 + e5) + (e6 + e7));
        s = fmaf(s, fast_exp2(m2e - nm), chunk);
        m2e = nm;

        pa += 16 * (DFEA / 8);   // 16 target rows ahead
        pm += 4;                 // 16 logw floats ahead
    }

    // merge the two lane-halves that share a src column (lane ^ 16)
    float mo = __shfl_xor(m2e, 16, 32);
    float so = __shfl_xor(s,   16, 32);
    float Mx = fmaxf(m2e, mo);
    float S  = s * fast_exp2(m2e - Mx) + so * fast_exp2(mo - Mx);

    // L = logsumexp over all targets (convert back from base-2 units)
    float L = (Mx + __builtin_amdgcn_logf(S)) * INV_L2E;

    if (lane < 16) {
        int   lbl = src_labels[col];
        float cnt = (float)ccount[lbl];
        contribArr[col] = msumArr[col] - cnt * L;
    }
}

// fixed-order final reduction: sum contribs; P = sum_c counts[c]*ccount[c]
__global__ void k_finred(const float* __restrict__ contrib,
                         const int* __restrict__ counts,
                         const int* __restrict__ ccount,
                         float* __restrict__ out, int Bs, int C) {
    __shared__ float sm[256];
    float t = 0.0f;
    for (int j = threadIdx.x; j < Bs; j += 256) t += contrib[j];
    sm[threadIdx.x] = t;
    __syncthreads();
    for (int st = 128; st > 0; st >>= 1) {
        if (threadIdx.x < st) sm[threadIdx.x] += sm[threadIdx.x + st];
        __syncthreads();
    }
    if (threadIdx.x == 0) {
        long long p = 0;
        for (int c = 0; c < C; ++c) p += (long long)counts[c] * (long long)ccount[c];
        out[0] = sm[0] / (-(float)Bs * (float)p);
    }
}

// ---- launcher -------------------------------------------------------------

extern "C" void kernel_launch(void* const* d_in, const int* in_sizes, int n_in,
                              void* d_out, int out_size, void* d_ws, size_t ws_size,
                              hipStream_t stream) {
    const float* src_feas   = (const float*)d_in[0];
    const int*   src_labels = (const int*)  d_in[1];
    const float* tgt_feas   = (const float*)d_in[2];
    const float* tgt_logits = (const float*)d_in[3];
    float* out = (float*)d_out;

    const int Bs = in_sizes[1];                 // 8192
    const int Bt = in_sizes[2] / DFEA;          // 8192
    const int C  = in_sizes[3] / Bt;            // 10

    // workspace layout
    char* ws = (char*)d_ws;
    size_t off = 0;
    _Float16* src16 = (_Float16*)(ws + off); off += (size_t)Bs * DFEA * sizeof(_Float16);
    _Float16* tgt16 = (_Float16*)(ws + off); off += (size_t)Bt * DFEA * sizeof(_Float16);
    off = (off + 255) & ~(size_t)255;
    float* metaLw  = (float*)(ws + off);     off += (size_t)Bt * sizeof(float);
    float* msumArr = (float*)(ws + off);     off += (size_t)Bs * sizeof(float);
    float* contrib = (float*)(ws + off);     off += (size_t)Bs * sizeof(float);
    int*   cls     = (int*)(ws + off);       off += (size_t)Bt * sizeof(int);
    float* tsum    = (float*)(ws + off);     off += (size_t)C * DFEA * sizeof(float);
    off = (off + 255) & ~(size_t)255;
    int* counts  = (int*)(ws + off);         off += 64;
    int* ccount  = (int*)(ws + off);         off += 64;

    k_zero<<<1, 32, 0, stream>>>(counts, ccount);
    k_counts<<<(Bs + 255) / 256, 256, 0, stream>>>(src_labels, counts, Bs);
    k_cls<<<(Bt + 255) / 256, 256, 0, stream>>>(tgt_logits, cls, ccount, Bt, C);
    k_cvt<<<(Bs * DFEA + 255) / 256, 256, 0, stream>>>(src_feas, src16, Bs * DFEA);
    k_cvt<<<(Bt * DFEA + 255) / 256, 256, 0, stream>>>(tgt_feas, tgt16, Bt * DFEA);
    k_metaLw<<<(Bt + 255) / 256, 256, 0, stream>>>(cls, counts, metaLw, Bt);
    k_tsum<<<(C * DFEA + 255) / 256, 256, 0, stream>>>(tgt_feas, cls, tsum, Bt, C);
    k_msum<<<(Bs + 255) / 256, 256, 0, stream>>>(src_feas, src_labels, tsum, msumArr, Bs);

    // 1 wave per 16 src rows: Bs/16 waves, 8 waves (256 threads) per block
    int waves  = Bs / 16;
    int blocks = waves / 8;
    k_main<<<blocks, 256, 0, stream>>>(src16, tgt16, metaLw, src_labels,
                                       msumArr, ccount, contrib, Bt);

    k_finred<<<1, 256, 0, stream>>>(contrib, counts, ccount, out, Bs, C);
}
